// LSTM_CRF_Model_9826885173326
// MI455X (gfx1250) — matrix operand order repaired and checked by reading, not compiled, not run on hardware
//
#include <hip/hip_runtime.h>

// ---------------------------------------------------------------------------
// LSTM + CRF for MI455X (gfx1250), wave32, WMMA f16 16x16x32,
// async global->LDS staging (ASYNCcnt) for weight tiles.
// ---------------------------------------------------------------------------

typedef __attribute__((ext_vector_type(16))) _Float16 v16h;
typedef __attribute__((ext_vector_type(8)))  _Float16 v8h;
typedef __attribute__((ext_vector_type(8)))  float    v8f;

#define S_LEN   512
#define B_SZ    64
#define E_DIM   256
#define H_DIM   512
#define T_TAGS  9
#define G4H     2048
#define NWG_REC 8

// ---- CDNA5 async global->LDS copy (GLOBAL_LOAD_ASYNC_TO_LDS_B128, GV mode) -
// vdst = per-lane LDS byte offset (low 32 bits of generic shared pointer),
// vaddr = 64-bit global address. Tracked by ASYNCcnt.
__device__ __forceinline__ void async_copy16(void* lds_ptr, const void* gptr) {
  unsigned l = (unsigned)(uintptr_t)lds_ptr;
  asm volatile("global_load_async_to_lds_b128 %0, %1, off"
               :: "v"(l), "v"(gptr) : "memory");
}
__device__ __forceinline__ void wait_async0() {
  asm volatile("s_wait_asynccnt 0x0" ::: "memory");
}

// ---- WMMA fragment helpers (layouts per CDNA5 ISA 7.12.2) ------------------

__device__ __forceinline__ v16h mk16(v8h lo, v8h hi) {
  v16h r;
#pragma unroll
  for (int i = 0; i < 8; ++i) { r[i] = lo[i]; r[i + 8] = hi[i]; }
  return r;
}

// A 16x32 f16: lane L -> row m = L&15, khigh = L>>4.
// halves j=0..7  hold k = kbase + 8*khigh + j
// halves j=8..15 hold k = kbase + 16 + 8*khigh + (j-8)
__device__ __forceinline__ v16h loadA(const _Float16* base, int lda, int m,
                                      int kbase, int khi) {
  const _Float16* p = base + (size_t)m * lda + kbase + 8 * khi;
  v8h lo = *reinterpret_cast<const v8h*>(p);
  v8h h2 = *reinterpret_cast<const v8h*>(p + 16);
  return mk16(lo, h2);
}

// B 32x16 f16: lane L -> col n = L&15, khigh = L>>4; same k layout as A:
// halves j=0..7 hold k = kbase + 8*khigh + j, j=8..15 hold k = kbase + 16 + 8*khigh + (j-8).
// LDS stores B^T as [n][k] rows, padded pitch.
__device__ __forceinline__ v16h loadB_lds(const _Float16* base, int pitch, int n,
                                          int kbase, int khi) {
  const _Float16* p = base + (size_t)n * pitch + kbase + 8 * khi;
  v8h lo = *reinterpret_cast<const v8h*>(p);
  v8h h2 = *reinterpret_cast<const v8h*>(p + 16);
  return mk16(lo, h2);
}

__device__ __forceinline__ v8f wmma32(v16h a, v16h b, v8f c) {
  return __builtin_amdgcn_wmma_f32_16x16x32_f16(false, a, false, b,
                                                (short)0, c, false, false);
}

__device__ __forceinline__ float sigm(float x) { return 1.0f / (1.0f + expf(-x)); }

// ---------------------------------------------------------------------------
// K0: zero h_{-1} and the grid-barrier counter (graph-replay safe).
// ---------------------------------------------------------------------------
__global__ void k_init(unsigned* bar, _Float16* hs0) {
  int i = blockIdx.x * blockDim.x + threadIdx.x;
  if (i < B_SZ * H_DIM) hs0[i] = (_Float16)0.0f;
  if (i == 0) *bar = 0u;
}

// ---------------------------------------------------------------------------
// K1: weights fp32 -> f16, bias sum.
// ---------------------------------------------------------------------------
__global__ void k_cvt(const float* Wih, const float* Whh, const float* bih,
                      const float* bhh, _Float16* Wih_h, _Float16* Whh_h,
                      float* bsum) {
  int i = blockIdx.x * blockDim.x + threadIdx.x;
  if (i < G4H * E_DIM) Wih_h[i] = (_Float16)Wih[i];
  if (i < G4H * H_DIM) Whh_h[i] = (_Float16)Whh[i];
  if (i < G4H) bsum[i] = bih[i] + bhh[i];
}

// ---------------------------------------------------------------------------
// K2: embedding gather, scale sqrt(E)=16, f16.
// ---------------------------------------------------------------------------
__global__ void k_embed(const int* x, const float* emb, _Float16* embeds) {
  int row = blockIdx.x;          // 0..S*B-1
  int col = threadIdx.x;         // 0..255
  int tok = x[row];
  embeds[(size_t)row * E_DIM + col] = (_Float16)(emb[(size_t)tok * E_DIM + col] * 16.0f);
}

// ---------------------------------------------------------------------------
// K3: input projection GEMM, M=32768, N=2048, K=256, WMMA f16.
// Grid (512, 8); per WG a 64x256 tile; W_ih slice async-staged into LDS.
// ---------------------------------------------------------------------------
__global__ void k_gemm_in(const _Float16* embeds, const _Float16* Wih_h,
                          const float* bsum, _Float16* gates) {
  extern __shared__ _Float16 lB[];   // 256 rows x 264 halves (4-bank stagger)
  const int PITCH = 264;
  const int nbase = blockIdx.y * 256;
  {
    int trow = threadIdx.x;          // one W_ih row per thread
    const _Float16* src = Wih_h + (size_t)(nbase + trow) * E_DIM;
    _Float16* dst = lB + (size_t)trow * PITCH;
#pragma unroll
    for (int j = 0; j < E_DIM / 8; ++j)
      async_copy16(dst + j * 8, src + j * 8);
  }
  wait_async0();
  __syncthreads();

  const int w = threadIdx.x >> 5, lane = threadIdx.x & 31;
  const int khi = lane >> 4, ln = lane & 15;
  const int mh = w >> 2, nq = w & 3;
  const int row0 = blockIdx.x * 64;

  v8f acc[2][4];
#pragma unroll
  for (int mi = 0; mi < 2; ++mi)
#pragma unroll
    for (int j = 0; j < 4; ++j) {
      float bs = bsum[nbase + (nq * 4 + j) * 16 + ln];
#pragma unroll
      for (int r = 0; r < 8; ++r) acc[mi][j][r] = bs;
    }

#pragma unroll 1
  for (int kk = 0; kk < E_DIM / 32; ++kk) {
    v16h a[2];
#pragma unroll
    for (int mi = 0; mi < 2; ++mi)
      a[mi] = loadA(embeds, E_DIM, row0 + (2 * mh + mi) * 16 + ln, kk * 32, khi);
#pragma unroll
    for (int j = 0; j < 4; ++j) {
      v16h b = loadB_lds(lB, PITCH, (nq * 4 + j) * 16 + ln, kk * 32, khi);
#pragma unroll
      for (int mi = 0; mi < 2; ++mi) acc[mi][j] = wmma32(a[mi], b, acc[mi][j]);
    }
  }

#pragma unroll
  for (int mi = 0; mi < 2; ++mi)
#pragma unroll
    for (int j = 0; j < 4; ++j) {
      int n_gl = nbase + (nq * 4 + j) * 16 + ln;
#pragma unroll
      for (int r = 0; r < 8; ++r) {
        int m = row0 + (2 * mh + mi) * 16 + r + 8 * khi;
        gates[(size_t)m * G4H + n_gl] = (_Float16)acc[mi][j][r];
      }
    }
}

// ---------------------------------------------------------------------------
// K4: persistent recurrent LSTM. 8 WGs x 256 threads; WG g owns hidden units
// [g*64, g*64+64) for all four gates. Its W_hh slice (256x512 f16, 266KB
// padded) is async-staged into LDS once and stays resident for all 512 steps.
// Cell state c lives in VGPRs. h is exchanged via global hs[] with an atomic
// grid barrier per step (8 WGs are always co-resident).
// ---------------------------------------------------------------------------
__global__ void k_lstm(const _Float16* __restrict__ Whh_h,
                       const _Float16* __restrict__ gates,
                       _Float16* __restrict__ hs, unsigned* bar) {
  extern __shared__ _Float16 lW[];   // 256 rows x 520 halves
  const int PITCH = 520;
  const int g = blockIdx.x;          // 0..7
  const int u0 = g * 64;
  {
    int n = threadIdx.x;                               // slice col 0..255
    int r = (n >> 6) * H_DIM + u0 + (n & 63);          // gate block * H + unit
    const _Float16* src = Whh_h + (size_t)r * H_DIM;
    _Float16* dst = lW + (size_t)n * PITCH;
#pragma unroll
    for (int j = 0; j < H_DIM / 8; ++j)
      async_copy16(dst + j * 8, src + j * 8);
  }
  wait_async0();
  __syncthreads();

  const int w = threadIdx.x >> 5, lane = threadIdx.x & 31;
  const int khi = lane >> 4, ln = lane & 15;
  const int mh = w >> 2;             // 2 M-tiles per wave
  const int t = w & 3;               // unit sub-tile; wave holds i/f/g/o for same units
  const int ucol = u0 + t * 16 + ln; // this lane's hidden unit

  float cst[2][8];
#pragma unroll
  for (int mi = 0; mi < 2; ++mi)
#pragma unroll
    for (int r = 0; r < 8; ++r) cst[mi][r] = 0.0f;

  for (int s = 0; s < S_LEN; ++s) {
    const _Float16* hprev = hs + (size_t)s * (B_SZ * H_DIM);
    _Float16* hnext = hs + (size_t)(s + 1) * (B_SZ * H_DIM);
    const _Float16* grow = gates + (size_t)s * B_SZ * G4H;

    v8f acc[2][4];
#pragma unroll
    for (int mi = 0; mi < 2; ++mi)
#pragma unroll
      for (int gg = 0; gg < 4; ++gg) {
        int rcol = gg * H_DIM + ucol;
#pragma unroll
        for (int r = 0; r < 8; ++r) {
          int m = (2 * mh + mi) * 16 + r + 8 * khi;
          acc[mi][gg][r] = (float)grow[(size_t)m * G4H + rcol];
        }
      }

#pragma unroll 1
    for (int kk = 0; kk < H_DIM / 32; ++kk) {
      v16h a[2];
#pragma unroll
      for (int mi = 0; mi < 2; ++mi)
        a[mi] = loadA(hprev, H_DIM, (2 * mh + mi) * 16 + ln, kk * 32, khi);
#pragma unroll
      for (int gg = 0; gg < 4; ++gg) {
        v16h b = loadB_lds(lW, PITCH, (gg * 4 + t) * 16 + ln, kk * 32, khi);
#pragma unroll
        for (int mi = 0; mi < 2; ++mi) acc[mi][gg] = wmma32(a[mi], b, acc[mi][gg]);
      }
    }

    // Cell update entirely in registers: this lane has i/f/g/o for its unit.
#pragma unroll
    for (int mi = 0; mi < 2; ++mi)
#pragma unroll
      for (int r = 0; r < 8; ++r) {
        float iv = sigm(acc[mi][0][r]);
        float fv = sigm(acc[mi][1][r]);
        float gv = tanhf(acc[mi][2][r]);
        float ov = sigm(acc[mi][3][r]);
        float c = fv * cst[mi][r] + iv * gv;
        cst[mi][r] = c;
        float h = ov * tanhf(c);
        int m = (2 * mh + mi) * 16 + r + 8 * khi;
        hnext[(size_t)m * H_DIM + ucol] = (_Float16)h;
      }

    if (s + 1 < S_LEN)   // warm L2 for next step's gate slice
      __builtin_prefetch(gates + (size_t)(s + 1) * B_SZ * G4H + (size_t)threadIdx.x * 64, 0, 1);

    // device-wide step barrier
    __threadfence();
    __syncthreads();
    if (threadIdx.x == 0) {
      atomicAdd(bar, 1u);
      unsigned target = (unsigned)(NWG_REC * (s + 1));
      while (atomicAdd(bar, 0u) < target) { __builtin_amdgcn_s_sleep(2); }
    }
    __syncthreads();
    __threadfence();
  }
}

// ---------------------------------------------------------------------------
// K5: logits + log_softmax -> emissions. One thread per (s,b); T=9 tiny.
// ---------------------------------------------------------------------------
__global__ void k_emis(const _Float16* hs, const float* Wout, const float* bout,
                       float* emis) {
  int idx = blockIdx.x * blockDim.x + threadIdx.x;
  if (idx >= S_LEN * B_SZ) return;
  int s = idx / B_SZ, b = idx % B_SZ;
  const _Float16* h = hs + (size_t)(s + 1) * (B_SZ * H_DIM) + (size_t)b * H_DIM;
  float lg[T_TAGS];
#pragma unroll
  for (int t = 0; t < T_TAGS; ++t) lg[t] = bout[t];
  for (int k = 0; k < H_DIM; ++k) {
    float hk = (float)h[k];
#pragma unroll
    for (int t = 0; t < T_TAGS; ++t) lg[t] += hk * Wout[t * H_DIM + k];
  }
  float mx = lg[0];
#pragma unroll
  for (int t = 1; t < T_TAGS; ++t) mx = fmaxf(mx, lg[t]);
  float se = 0.f;
#pragma unroll
  for (int t = 0; t < T_TAGS; ++t) se += expf(lg[t] - mx);
  float lse = mx + logf(se);
  float* e = emis + (size_t)idx * T_TAGS;
#pragma unroll
  for (int t = 0; t < T_TAGS; ++t) e[t] = lg[t] - lse;
}

// ---------------------------------------------------------------------------
// K6: CRF negative log-likelihood (scalar). One lane per batch element.
// ---------------------------------------------------------------------------
__global__ void k_crf(const float* emis, const int* bio, const int* x,
                      const float* start_t, const float* end_t,
                      const float* trans, float* out) {
  __shared__ float red[B_SZ];
  int b = threadIdx.x;
  int cnt = 0;
  for (int s = 0; s < S_LEN; ++s) cnt += (x[s * B_SZ + b] != 0) ? 1 : 0;
  int seq_end = cnt - 1;
  int last_tag = bio[seq_end * B_SZ + b];

  int tprev = bio[b];
  float num = start_t[tprev] + emis[(size_t)b * T_TAGS + tprev];

  float score[T_TAGS];
#pragma unroll
  for (int t = 0; t < T_TAGS; ++t) score[t] = start_t[t] + emis[(size_t)b * T_TAGS + t];

  for (int s = 1; s < S_LEN; ++s) {
    const float* e = emis + (size_t)(s * B_SZ + b) * T_TAGS;
    float mk = (x[s * B_SZ + b] != 0) ? 1.0f : 0.0f;
    int tcur = bio[s * B_SZ + b];
    num += (e[tcur] + trans[tprev * T_TAGS + tcur]) * mk;
    tprev = tcur;
    if (mk > 0.f) {
      float ns[T_TAGS];
#pragma unroll
      for (int t2 = 0; t2 < T_TAGS; ++t2) {
        float mx = -3.0e38f;
#pragma unroll
        for (int t1 = 0; t1 < T_TAGS; ++t1)
          mx = fmaxf(mx, score[t1] + trans[t1 * T_TAGS + t2]);
        float ss = 0.f;
#pragma unroll
        for (int t1 = 0; t1 < T_TAGS; ++t1)
          ss += expf(score[t1] + trans[t1 * T_TAGS + t2] - mx);
        ns[t2] = mx + logf(ss) + e[t2];
      }
#pragma unroll
      for (int t = 0; t < T_TAGS; ++t) score[t] = ns[t];
    }
  }
  num += end_t[last_tag];
  float mx = -3.0e38f;
#pragma unroll
  for (int t = 0; t < T_TAGS; ++t) mx = fmaxf(mx, score[t] + end_t[t]);
  float ss = 0.f;
#pragma unroll
  for (int t = 0; t < T_TAGS; ++t) ss += expf(score[t] + end_t[t] - mx);
  float norm = mx + logf(ss);

  red[b] = num - norm;
  __syncthreads();
  if (b == 0) {
    float tot = 0.f;
    for (int i = 0; i < B_SZ; ++i) tot += red[i];
    out[0] = -tot;
  }
}

// ---------------------------------------------------------------------------
// K7: Viterbi decode. One lane per batch element; history in scratch.
// ---------------------------------------------------------------------------
__global__ void k_vit(const float* emis, const float* start_t, const float* end_t,
                      const float* trans, unsigned char* hist, float* out) {
  int b = threadIdx.x;
  float score[T_TAGS];
#pragma unroll
  for (int t = 0; t < T_TAGS; ++t) score[t] = start_t[t] + emis[(size_t)b * T_TAGS + t];

  for (int s = 1; s < S_LEN; ++s) {
    const float* e = emis + (size_t)(s * B_SZ + b) * T_TAGS;
    float ns[T_TAGS]; int bi[T_TAGS];
#pragma unroll
    for (int t2 = 0; t2 < T_TAGS; ++t2) {
      float best = score[0] + trans[t2];
      int idx = 0;
#pragma unroll
      for (int t1 = 1; t1 < T_TAGS; ++t1) {
        float v = score[t1] + trans[t1 * T_TAGS + t2];
        if (v > best) { best = v; idx = t1; }
      }
      ns[t2] = best + e[t2];
      bi[t2] = idx;
    }
#pragma unroll
    for (int t = 0; t < T_TAGS; ++t) {
      score[t] = ns[t];
      hist[((size_t)s * B_SZ + b) * T_TAGS + t] = (unsigned char)bi[t];
    }
  }
  float best = score[0] + end_t[0]; int last = 0;
#pragma unroll
  for (int t = 1; t < T_TAGS; ++t) {
    float v = score[t] + end_t[t];
    if (v > best) { best = v; last = t; }
  }
  int cur = last;
  for (int s = S_LEN - 1; s >= 1; --s) {
    out[1 + s * B_SZ + b] = (float)cur;
    cur = hist[((size_t)s * B_SZ + b) * T_TAGS + cur];
  }
  out[1 + b] = (float)cur;
}

// ---------------------------------------------------------------------------
extern "C" void kernel_launch(void* const* d_in, const int* in_sizes, int n_in,
                              void* d_out, int out_size, void* d_ws, size_t ws_size,
                              hipStream_t stream) {
  (void)in_sizes; (void)n_in; (void)out_size; (void)ws_size;
  const int*   x      = (const int*)d_in[0];
  const int*   bio    = (const int*)d_in[1];
  const float* emb    = (const float*)d_in[2];
  const float* W_ih   = (const float*)d_in[3];
  const float* W_hh   = (const float*)d_in[4];
  const float* b_ih   = (const float*)d_in[5];
  const float* b_hh   = (const float*)d_in[6];
  const float* W_out  = (const float*)d_in[7];
  const float* b_out  = (const float*)d_in[8];
  const float* startt = (const float*)d_in[9];
  const float* endt   = (const float*)d_in[10];
  const float* trans  = (const float*)d_in[11];
  float* out = (float*)d_out;

  char* ws = (char*)d_ws;
  size_t off = 0;
  auto take = [&](size_t bytes) -> size_t {
    size_t r = off; off += (bytes + 255) & ~(size_t)255; return r;
  };
  unsigned*      bar    = (unsigned*)(ws + take(256));
  _Float16*      Wih_h  = (_Float16*)(ws + take((size_t)G4H * E_DIM * 2));
  _Float16*      Whh_h  = (_Float16*)(ws + take((size_t)G4H * H_DIM * 2));
  float*         bsum   = (float*)(ws + take((size_t)G4H * 4));
  _Float16*      embeds = (_Float16*)(ws + take((size_t)S_LEN * B_SZ * E_DIM * 2));
  _Float16*      gates  = (_Float16*)(ws + take((size_t)S_LEN * B_SZ * G4H * 2));
  _Float16*      hs     = (_Float16*)(ws + take((size_t)(S_LEN + 1) * B_SZ * H_DIM * 2));
  float*         emis   = (float*)(ws + take((size_t)S_LEN * B_SZ * T_TAGS * 4));
  unsigned char* hist   = (unsigned char*)(ws + take((size_t)S_LEN * B_SZ * T_TAGS));

  k_init<<<dim3((B_SZ * H_DIM + 255) / 256), 256, 0, stream>>>(bar, hs);
  k_cvt<<<dim3((G4H * H_DIM + 255) / 256), 256, 0, stream>>>(
      W_ih, W_hh, b_ih, b_hh, Wih_h, Whh_h, bsum);
  k_embed<<<dim3(S_LEN * B_SZ), 256, 0, stream>>>(x, emb, embeds);
  k_gemm_in<<<dim3((S_LEN * B_SZ) / 64, G4H / 256), 256,
              256 * 264 * sizeof(_Float16), stream>>>(embeds, Wih_h, bsum, gates);
  k_lstm<<<dim3(NWG_REC), 256, 256 * 520 * sizeof(_Float16), stream>>>(
      Whh_h, gates, hs, bar);
  k_emis<<<dim3((S_LEN * B_SZ + 255) / 256), 256, 0, stream>>>(hs, W_out, b_out, emis);
  k_crf<<<dim3(1), B_SZ, 0, stream>>>(emis, bio, x, startt, endt, trans, out);
  k_vit<<<dim3(1), B_SZ, 0, stream>>>(emis, startt, endt, trans, hist, out);
}
